// CrossAttentionGates_8521215115946
// MI455X (gfx1250) — matrix operand order repaired and checked by reading, not compiled, not run on hardware
//
#include <hip/hip_runtime.h>
#include <hip/hip_bf16.h>

// ---------------------------------------------------------------------------
// CDNA5 (gfx1250) graph cross-attention block.
// Prep: one-time f32->bf16 conversion of activations + transposed bf16 weights.
// GEMMs: both LDS tiles arrive via TENSOR_LOAD_TO_LDS (TDM, hardware row
// padding -> conflict-free pitch 136), compute is pure ds_load_b128 +
// v_wmma_f32_16x16x32_bf16. Attention: L2-resident gather/scatter passes with
// wave32 shuffles and float atomics.
// ---------------------------------------------------------------------------

typedef __attribute__((ext_vector_type(16))) __bf16 v16bf;
typedef __attribute__((ext_vector_type(8)))  float  v8f;
typedef unsigned int u32x4 __attribute__((ext_vector_type(4)));
typedef int          i32x8 __attribute__((ext_vector_type(8)));
typedef int          i32x4 __attribute__((ext_vector_type(4)));

union BF16x16 {
    v16bf v;
    unsigned short u[16];
    uint4 q[2];
};

__device__ __forceinline__ unsigned short f32_to_bf16(float f) {
    unsigned int u = __float_as_uint(f);
    u += 0x7FFFu + ((u >> 16) & 1u);      // round-to-nearest-even
    return (unsigned short)(u >> 16);
}

// monotonic float<->uint transform for atomicMax-based segment max
__device__ __forceinline__ unsigned int f2ord(float x) {
    unsigned int u = __float_as_uint(x);
    return (u & 0x80000000u) ? ~u : (u | 0x80000000u);
}
__device__ __forceinline__ float ord2f(unsigned int k) {
    return (k & 0x80000000u) ? __uint_as_float(k & 0x7FFFFFFFu)
                             : __uint_as_float(~k);
}

__device__ __forceinline__ float sigmoidf_(float x) {
    return 1.0f / (1.0f + __expf(-x));
}

#define D128      128
#define HEADS     4
#define SCALE_H   0.17677669529663687f   // 1/sqrt(32)
#define TP        136                    // LDS pitch (bf16) after TDM padding

// 2D bf16 tile load via Tensor Data Mover.
// Row = 128 bf16 = 256B = 64 dwords -> pad_interval code 5; +4 dwords pad
// (code 3) -> LDS pitch 136 bf16 (68 dwords, lane stride 4 banks).
__device__ __forceinline__ void tdm_load_2d_bf16(
    unsigned ldsoff, const void* gaddr, unsigned dim1, unsigned tile1)
{
    unsigned long long ga = (unsigned long long)gaddr;
    u32x4 g0;
    g0[0] = 1u;                                 // count=1, user descriptor
    g0[1] = ldsoff;                             // lds_addr (byte offset)
    g0[2] = (unsigned)(ga & 0xFFFFFFFFull);     // global_addr[31:0]
    g0[3] = (unsigned)((ga >> 32) & 0x01FFFFFFull) | (2u << 30); // [56:32]|type=2

    i32x8 g1;
    g1[0] = (1 << 16)      // data_size = 2 bytes (bf16)
          | (1 << 20)      // pad_enable
          | (5 << 22)      // pad_interval: 64 dwords (one 128-bf16 row)
          | (3 << 25);     // pad_amount: 4 dwords
    g1[1] = (int)(128u << 16);                              // tensor_dim0 = 128
    g1[2] = (int)((dim1 & 0xFFFFu) << 16);                  // tensor_dim1 lo
    g1[3] = (int)(((dim1 >> 16) & 0xFFFFu) | (128u << 16)); // dim1 hi | tile_dim0
    g1[4] = (int)tile1;                                     // tile_dim1, tile_dim2=0
    g1[5] = 128;                                            // tensor_dim0_stride
    g1[6] = 0;
    g1[7] = 0;

    i32x4 z4 = {0, 0, 0, 0};
    i32x8 z8 = {0, 0, 0, 0, 0, 0, 0, 0};
    __builtin_amdgcn_tensor_load_to_lds(g0, g1, z4, z4, z8, 0);
}

// ---------------------------------------------------------------------------
// Prep: f32 -> bf16 row-major (activations), and W -> bf16 W^T (weights).
// ---------------------------------------------------------------------------
__global__ void cvt_bf16_kernel(const float* __restrict__ in,
                                unsigned short* __restrict__ out, long n)
{
    long i = ((long)blockIdx.x * blockDim.x + threadIdx.x) * 4;
    if (i < n) {
        float4 v = *(const float4*)(in + i);
        ushort4 o;
        o.x = f32_to_bf16(v.x); o.y = f32_to_bf16(v.y);
        o.z = f32_to_bf16(v.z); o.w = f32_to_bf16(v.w);
        *(ushort4*)(out + i) = o;
    }
}

__global__ void transpose_w_bf16_kernel(const float* __restrict__ W,
                                        unsigned short* __restrict__ Wt)
{
    int i = blockIdx.x * blockDim.x + threadIdx.x;   // 16384 threads
    int k = i >> 7, c = i & 127;
    Wt[c * D128 + k] = f32_to_bf16(W[i]);
}

// ---------------------------------------------------------------------------
// GEMM: Y[rows x 128] = X[rows x 128] @ W[128 x 128] + bias (bf16 WMMA).
// Block = 128 threads (4 waves), 64 rows/block. Both tiles DMA'd by TDM.
// ---------------------------------------------------------------------------
__global__ __launch_bounds__(128) void gemm128_wmma_kernel(
    const unsigned short* __restrict__ Xbf,  // rows x 128 bf16 row-major
    const unsigned short* __restrict__ Wt,   // 128 x 128 bf16, [col][k]
    const float* __restrict__ bias, float* __restrict__ Y, long rows)
{
    __shared__ __align__(16) unsigned short sA[64 * TP];    // A tile (bf16)
    __shared__ __align__(16) unsigned short sW[D128 * TP];  // W^T (bf16)

    const int  t       = threadIdx.x;
    const long rowBase = (long)blockIdx.x * 64;

    if ((t >> 5) == 0) {                      // wave 0 issues both DMAs
        long rem = rows - rowBase;            // TDM zero-fills rows >= rem
        unsigned dim1 = (unsigned)(rem > 0 ? rem : 0);
        tdm_load_2d_bf16((unsigned)(unsigned long long)(void*)&sA[0],
                         Xbf + rowBase * D128, dim1, 64u);
        tdm_load_2d_bf16((unsigned)(unsigned long long)(void*)&sW[0],
                         Wt, 128u, 128u);
    }
    __builtin_amdgcn_s_wait_tensorcnt(0);
    __syncthreads();

    const int lane = t & 31;
    const int wave = t >> 5;
    const int half = lane >> 4;   // K-half select per 16-bit WMMA layout
    const int m    = lane & 15;   // M (A) / N (B,D) index
    const int r0   = wave * 16;

    // A fragments: VGPR0..3 -> K {0..7 | 8..15}, VGPR4..7 -> K {16..23 | 24..31}
    BF16x16 afr[4];
    const unsigned short* pa = &sA[(r0 + m) * TP];
#pragma unroll
    for (int kb = 0; kb < 4; ++kb) {
        int k0 = kb * 32 + half * 8;
        afr[kb].q[0] = *(const uint4*)(pa + k0);
        afr[kb].q[1] = *(const uint4*)(pa + k0 + 16);
    }

    const bool fullTile = (rowBase + 64 <= rows);

#pragma unroll
    for (int ct = 0; ct < 8; ++ct) {
        const int col = ct * 16 + m;
        v8f acc = {};
#pragma unroll
        for (int kb = 0; kb < 4; ++kb) {
            BF16x16 bfr;
            const unsigned short* pw = &sW[col * TP + kb * 32 + half * 8];
            bfr.q[0] = *(const uint4*)(pw);      // K kk..kk+7
            bfr.q[1] = *(const uint4*)(pw + 16); // K kk+16..kk+23
            acc = __builtin_amdgcn_wmma_f32_16x16x32_bf16(
                false, afr[kb].v, false, bfr.v, (short)0, acc, false, false);
        }
        const float b = bias[col];
        float* yb = Y + (rowBase + r0 + half * 8) * D128 + col;
        if (fullTile) {                          // uniform fast path: no exec churn
#pragma unroll
            for (int i = 0; i < 8; ++i)          // D: VGPR i -> M = i + 8*half
                yb[(long)i * D128] = acc[i] + b;
        } else {
#pragma unroll
            for (int i = 0; i < 8; ++i) {
                long row = rowBase + r0 + i + half * 8;
                if (row < rows) Y[row * D128 + col] = acc[i] + b;
            }
        }
    }
}

// ---------------------------------------------------------------------------
// Init: zero node accumulator, segment max (ordered-uint 0 = -inf) and sums.
// ---------------------------------------------------------------------------
__global__ void init_kernel(unsigned int* __restrict__ maxb,
                            float* __restrict__ sums,
                            float* __restrict__ acc, long N_)
{
    long i = (long)blockIdx.x * blockDim.x + threadIdx.x;
    if (i < N_ * D128) acc[i] = 0.0f;
    if (i < N_ * HEADS) { maxb[i] = 0u; sums[i] = 0.0f; }
}

// ---------------------------------------------------------------------------
// Node-attention pass 1: per-edge per-head scores + segment max (atomicMax).
// ---------------------------------------------------------------------------
__global__ __launch_bounds__(128) void node_score_kernel(
    const float* __restrict__ nq, const float* __restrict__ ekf,
    const int* __restrict__ ei, float* __restrict__ sS, float* __restrict__ sT,
    unsigned int* __restrict__ maxb, long E_)
{
    long e = (long)blockIdx.x * 4 + (threadIdx.x >> 5);
    if (e >= E_) return;
    const int lane = threadIdx.x & 31;
    const int h = lane >> 3, j = lane & 7;
    const int s = ei[e], tg = ei[E_ + e];
    const int off = h * 32 + j * 4;

    __builtin_prefetch(&nq[(long)tg * D128 + off], 0, 3);
    float4 qs = *(const float4*)&nq[(long)s  * D128 + off];
    float4 qt = *(const float4*)&nq[(long)tg * D128 + off];
    float4 kv = *(const float4*)&ekf[e * D128 + off];

    float ps = qs.x*kv.x + qs.y*kv.y + qs.z*kv.z + qs.w*kv.w;
    float pt = qt.x*kv.x + qt.y*kv.y + qt.z*kv.z + qt.w*kv.w;
#pragma unroll
    for (int mk = 1; mk <= 4; mk <<= 1) {
        ps += __shfl_xor(ps, mk, 32);
        pt += __shfl_xor(pt, mk, 32);
    }
    ps *= SCALE_H; pt *= SCALE_H;
    if (j == 0) {
        sS[e * HEADS + h] = ps;
        sT[e * HEADS + h] = pt;
        atomicMax(&maxb[(long)s  * HEADS + h], f2ord(ps));
        atomicMax(&maxb[(long)tg * HEADS + h], f2ord(pt));
    }
}

// ---------------------------------------------------------------------------
// Node-attention pass 2: exp(score - segmax), segment sums (atomicAdd).
// ---------------------------------------------------------------------------
__global__ void node_expsum_kernel(
    const int* __restrict__ ei, float* __restrict__ sS, float* __restrict__ sT,
    const unsigned int* __restrict__ maxb, float* __restrict__ sums, long E_)
{
    long idx = (long)blockIdx.x * blockDim.x + threadIdx.x;
    if (idx >= E_ * HEADS) return;
    long e = idx >> 2; int h = (int)(idx & 3);
    const int s = ei[e], tg = ei[E_ + e];
    float es = __expf(sS[idx] - ord2f(maxb[(long)s  * HEADS + h]));
    float et = __expf(sT[idx] - ord2f(maxb[(long)tg * HEADS + h]));
    sS[idx] = es; sT[idx] = et;
    atomicAdd(&sums[(long)s  * HEADS + h], es);
    atomicAdd(&sums[(long)tg * HEADS + h], et);
}

// ---------------------------------------------------------------------------
// Node-attention pass 3: scatter-add a*ev into node accumulator (L2-resident).
// ---------------------------------------------------------------------------
__global__ __launch_bounds__(256) void node_agg_kernel(
    const int* __restrict__ ei, const float* __restrict__ sS,
    const float* __restrict__ sT, const float* __restrict__ sums,
    const float* __restrict__ evf, float* __restrict__ acc, long E_)
{
    long e = (long)blockIdx.x * 2 + (threadIdx.x >> 7);
    if (e >= E_) return;
    const int d = threadIdx.x & 127, h = d >> 5;
    const int s = ei[e], tg = ei[E_ + e];
    float aS = sS[e * HEADS + h] / (sums[(long)s  * HEADS + h] + 1e-10f);
    float aT = sT[e * HEADS + h] / (sums[(long)tg * HEADS + h] + 1e-10f);
    float v = evf[e * D128 + d];
    atomicAdd(&acc[(long)s  * D128 + d], aS * v);
    atomicAdd(&acc[(long)tg * D128 + d], aT * v);
}

// ---------------------------------------------------------------------------
// Node residual + gate + LayerNorm. One wave per node (4 floats / lane).
// ---------------------------------------------------------------------------
__global__ __launch_bounds__(128) void node_ln_kernel(
    const float* __restrict__ nf, const float* __restrict__ acc,
    const float* __restrict__ gate, const float* __restrict__ lw,
    const float* __restrict__ lb, float* __restrict__ out, long N_)
{
    long n = (long)blockIdx.x * 4 + (threadIdx.x >> 5);
    if (n >= N_) return;
    const int lane = threadIdx.x & 31;
    const int d0 = lane * 4;
    const float g = sigmoidf_(gate[d0 >> 5]);

    float4 x = *(const float4*)&nf[n * D128 + d0];
    float4 a = *(const float4*)&acc[n * D128 + d0];
    x.x += g * a.x; x.y += g * a.y; x.z += g * a.z; x.w += g * a.w;

    float sum = x.x + x.y + x.z + x.w;
    float sq  = x.x*x.x + x.y*x.y + x.z*x.z + x.w*x.w;
#pragma unroll
    for (int mk = 1; mk <= 16; mk <<= 1) {
        sum += __shfl_xor(sum, mk, 32);
        sq  += __shfl_xor(sq,  mk, 32);
    }
    float mean = sum * (1.0f / 128.0f);
    float var  = sq * (1.0f / 128.0f) - mean * mean;
    float inv  = rsqrtf(var + 1e-5f);

    float4 w4 = *(const float4*)&lw[d0];
    float4 b4 = *(const float4*)&lb[d0];
    float4 y;
    y.x = (x.x - mean) * inv * w4.x + b4.x;
    y.y = (x.y - mean) * inv * w4.y + b4.y;
    y.z = (x.z - mean) * inv * w4.z + b4.z;
    y.w = (x.w - mean) * inv * w4.w + b4.w;
    *(float4*)&out[n * D128 + d0] = y;
}

// ---------------------------------------------------------------------------
// Edge update: 2-way softmax over (src,tgt), gate, residual, LayerNorm.
// ---------------------------------------------------------------------------
__global__ __launch_bounds__(128) void edge_update_kernel(
    const float* __restrict__ ef, const float* __restrict__ eqf,
    const float* __restrict__ nkf, const float* __restrict__ nvf,
    const int* __restrict__ ei, const float* __restrict__ gate,
    const float* __restrict__ lw, const float* __restrict__ lb,
    float* __restrict__ out, long E_)
{
    long e = (long)blockIdx.x * 4 + (threadIdx.x >> 5);
    if (e >= E_) return;
    const int lane = threadIdx.x & 31;
    const int h = lane >> 3;
    const int d0 = lane * 4;          // == h*32 + (lane&7)*4
    const int s = ei[e], tg = ei[E_ + e];

    __builtin_prefetch(&nvf[(long)s * D128 + d0], 0, 3);
    float4 q  = *(const float4*)&eqf[e * D128 + d0];
    float4 ks = *(const float4*)&nkf[(long)s  * D128 + d0];
    float4 kt = *(const float4*)&nkf[(long)tg * D128 + d0];

    float ps = q.x*ks.x + q.y*ks.y + q.z*ks.z + q.w*ks.w;
    float pt = q.x*kt.x + q.y*kt.y + q.z*kt.z + q.w*kt.w;
#pragma unroll
    for (int mk = 1; mk <= 4; mk <<= 1) {          // reduce within head group
        ps += __shfl_xor(ps, mk, 32);
        pt += __shfl_xor(pt, mk, 32);
    }
    ps *= SCALE_H; pt *= SCALE_H;
    float mm = fmaxf(ps, pt);
    float as = __expf(ps - mm), at = __expf(pt - mm);
    float dn = as + at + 1e-10f;
    float g  = sigmoidf_(gate[h]) / dn;

    float4 vs = *(const float4*)&nvf[(long)s  * D128 + d0];
    float4 vt = *(const float4*)&nvf[(long)tg * D128 + d0];
    float4 x  = *(const float4*)&ef[e * D128 + d0];
    x.x += g * (as * vs.x + at * vt.x);
    x.y += g * (as * vs.y + at * vt.y);
    x.z += g * (as * vs.z + at * vt.z);
    x.w += g * (as * vs.w + at * vt.w);

    float sum = x.x + x.y + x.z + x.w;
    float sq  = x.x*x.x + x.y*x.y + x.z*x.z + x.w*x.w;
#pragma unroll
    for (int mk = 1; mk <= 16; mk <<= 1) {
        sum += __shfl_xor(sum, mk, 32);
        sq  += __shfl_xor(sq,  mk, 32);
    }
    float mean = sum * (1.0f / 128.0f);
    float var  = sq * (1.0f / 128.0f) - mean * mean;
    float inv  = rsqrtf(var + 1e-5f);

    float4 w4 = *(const float4*)&lw[d0];
    float4 b4 = *(const float4*)&lb[d0];
    float4 y;
    y.x = (x.x - mean) * inv * w4.x + b4.x;
    y.y = (x.y - mean) * inv * w4.y + b4.y;
    y.z = (x.z - mean) * inv * w4.z + b4.z;
    y.w = (x.w - mean) * inv * w4.w + b4.w;
    *(float4*)&out[e * D128 + d0] = y;
}

// ---------------------------------------------------------------------------
extern "C" void kernel_launch(void* const* d_in, const int* in_sizes, int n_in,
                              void* d_out, int out_size, void* d_ws, size_t ws_size,
                              hipStream_t stream)
{
    const float* nf = (const float*)d_in[0];
    const float* ef = (const float*)d_in[1];
    const int*   ei = (const int*)d_in[2];
    const float* nq_w = (const float*)d_in[3];
    const float* nq_b = (const float*)d_in[4];
    const float* nk_w = (const float*)d_in[5];
    const float* nk_b = (const float*)d_in[6];
    const float* nv_w = (const float*)d_in[7];
    const float* nv_b = (const float*)d_in[8];
    const float* node_gate = (const float*)d_in[9];
    const float* eq_w = (const float*)d_in[10];
    const float* eq_b = (const float*)d_in[11];
    const float* ek_w = (const float*)d_in[12];
    const float* ek_b = (const float*)d_in[13];
    const float* ev_w = (const float*)d_in[14];
    const float* ev_b = (const float*)d_in[15];
    const float* edge_gate = (const float*)d_in[16];
    const float* ln_nw = (const float*)d_in[17];
    const float* ln_nb = (const float*)d_in[18];
    const float* ln_ew = (const float*)d_in[19];
    const float* ln_eb = (const float*)d_in[20];

    const long N = in_sizes[0] / D128;
    const long E = in_sizes[1] / D128;

    // Workspace layout (float units; all sub-offsets stay 8B-aligned)
    float* ws = (float*)d_ws;
    size_t o = 0;
    float* nq  = ws + o; o += (size_t)N * D128;   // node queries
    float* nkf = ws + o; o += (size_t)N * D128;   // node keys   (edge attn)
    float* nvf = ws + o; o += (size_t)N * D128;   // node values (edge attn)
    float* accN = ws + o; o += (size_t)N * D128;  // node cross accumulator
    float* ekf = ws + o; o += (size_t)E * D128;   // edge keys   (node attn)
    float* evf = ws + o; o += (size_t)E * D128;   // edge values (node attn)
    float* eqf = ws + o; o += (size_t)E * D128;   // edge queries
    float* sS  = ws + o; o += (size_t)E * HEADS;  // src scores -> exp
    float* sT  = ws + o; o += (size_t)E * HEADS;  // tgt scores -> exp
    float* sums = ws + o; o += (size_t)N * HEADS; // segment exp sums
    unsigned int* maxb = (unsigned int*)(ws + o); o += (size_t)N * HEADS;
    unsigned short* nf_bf = (unsigned short*)(ws + o); o += (size_t)N * D128 / 2;
    unsigned short* ef_bf = (unsigned short*)(ws + o); o += (size_t)E * D128 / 2;
    unsigned short* wt[6];
    for (int i = 0; i < 6; ++i) { wt[i] = (unsigned short*)(ws + o); o += (D128 * D128) / 2; }

    float* out_nodes = (float*)d_out;
    float* out_edges = (float*)d_out + (size_t)N * D128;

    // Zero accumulators / segment state.
    init_kernel<<<(unsigned)((N * D128 + 255) / 256), 256, 0, stream>>>(maxb, sums, accN, N);

    // One-time bf16 conversions: activations row-major, weights transposed.
    cvt_bf16_kernel<<<(unsigned)((N * D128 / 4 + 255) / 256), 256, 0, stream>>>(nf, nf_bf, N * D128);
    cvt_bf16_kernel<<<(unsigned)((E * D128 / 4 + 255) / 256), 256, 0, stream>>>(ef, ef_bf, E * D128);
    transpose_w_bf16_kernel<<<64, 256, 0, stream>>>(nq_w, wt[0]);
    transpose_w_bf16_kernel<<<64, 256, 0, stream>>>(nk_w, wt[1]);
    transpose_w_bf16_kernel<<<64, 256, 0, stream>>>(nv_w, wt[2]);
    transpose_w_bf16_kernel<<<64, 256, 0, stream>>>(eq_w, wt[3]);
    transpose_w_bf16_kernel<<<64, 256, 0, stream>>>(ek_w, wt[4]);
    transpose_w_bf16_kernel<<<64, 256, 0, stream>>>(ev_w, wt[5]);

    // Six 128x128 projections on WMMA (TDM-fed).
    const unsigned gN = (unsigned)((N + 63) / 64);
    const unsigned gE = (unsigned)((E + 63) / 64);
    gemm128_wmma_kernel<<<gN, 128, 0, stream>>>(nf_bf, wt[0], nq_b, nq,  N);
    gemm128_wmma_kernel<<<gE, 128, 0, stream>>>(ef_bf, wt[1], nk_b, ekf, E);
    gemm128_wmma_kernel<<<gE, 128, 0, stream>>>(ef_bf, wt[2], nv_b, evf, E);
    gemm128_wmma_kernel<<<gE, 128, 0, stream>>>(ef_bf, wt[3], eq_b, eqf, E);
    gemm128_wmma_kernel<<<gN, 128, 0, stream>>>(nf_bf, wt[4], ek_b, nkf, N);
    gemm128_wmma_kernel<<<gN, 128, 0, stream>>>(nf_bf, wt[5], ev_b, nvf, N);

    // Node-side segment softmax (3 passes) + LN.
    node_score_kernel<<<(unsigned)((E + 3) / 4), 128, 0, stream>>>(
        nq, ekf, ei, sS, sT, maxb, E);
    node_expsum_kernel<<<(unsigned)((E * HEADS + 255) / 256), 256, 0, stream>>>(
        ei, sS, sT, maxb, sums, E);
    node_agg_kernel<<<(unsigned)((E + 1) / 2), 256, 0, stream>>>(
        ei, sS, sT, sums, evf, accN, E);
    node_ln_kernel<<<(unsigned)((N + 3) / 4), 128, 0, stream>>>(
        nf, accN, node_gate, ln_nw, ln_nb, out_nodes, N);

    // Edge-side 2-way softmax + LN (single fused pass).
    edge_update_kernel<<<(unsigned)((E + 3) / 4), 128, 0, stream>>>(
        ef, eqf, nkf, nvf, ei, edge_gate, ln_ew, ln_eb, out_edges, E);
}
